// EmotionGNN_27573690040487
// MI455X (gfx1250) — compile-verified, matrix operand order checked by reading, not run on hardware
//
#include <hip/hip_runtime.h>
#include <hip/hip_bf16.h>

typedef __bf16 bf16;
typedef __attribute__((ext_vector_type(16))) __bf16 v16bf;
typedef __attribute__((ext_vector_type(8)))  float  v8f;

#define NODES 478
#define BATCH 1024

__device__ __forceinline__ float gelu_exact(float x) {
  return 0.5f * x * (1.0f + erff(x * 0.70710678118654752f));
}

// ---------------- weight convert: W (K x N) fp32 -> Wt (N x K) bf16 ----------------
__global__ void egnn_wconv_kernel(const float* __restrict__ w, bf16* __restrict__ wt,
                                  int K, int N) {
  int idx = blockIdx.x * blockDim.x + threadIdx.x;
  if (idx >= K * N) return;
  int k = idx / N, n = idx - k * N;
  wt[n * K + k] = (bf16)w[idx];
}

// ---------------- CSR build from edge_index (2,E): src=ei[0:E], dst=ei[E:2E] -------
__global__ void egnn_csr_kernel(const int* __restrict__ ei, int E,
                                int* deg, int* cursor, int* row_start, int* nbr,
                                float* invdeg) {
  int tid = threadIdx.x, nt = blockDim.x;
  for (int i = tid; i < NODES; i += nt) { deg[i] = 0; cursor[i] = 0; }
  __syncthreads();
  for (int e = tid; e < E; e += nt) atomicAdd(&deg[ei[e]], 1);
  __syncthreads();
  if (tid == 0) {
    int acc = 0;
    for (int i = 0; i < NODES; ++i) { row_start[i] = acc; acc += deg[i]; }
    row_start[NODES] = acc;
  }
  __syncthreads();
  for (int e = tid; e < E; e += nt) {
    int s = ei[e], d = ei[E + e];
    int pos = atomicAdd(&cursor[s], 1);
    nbr[row_start[s] + pos] = d;
  }
  __syncthreads();
  for (int i = tid; i < NODES; i += nt)
    invdeg[i] = 1.0f / ((float)deg[i] + 1e-6f);
}

// ---------------- input projection: (B,N,3) @ (3,128) + b -> bf16 ------------------
__global__ void egnn_inproj_kernel(const float* __restrict__ x,
                                   const float* __restrict__ w,
                                   const float* __restrict__ b,
                                   bf16* __restrict__ h) {
  int n = blockIdx.x, bb = blockIdx.y, f = threadIdx.x;  // 128 threads
  const float* xv = x + (size_t)bb * (NODES * 3) + n * 3;
  float acc = b[f] + xv[0] * w[f] + xv[1] * w[128 + f] + xv[2] * w[256 + f];
  h[((size_t)bb * NODES + n) * 128 + f] = (bf16)acc;
}

// ---------------- bf16 WMMA GEMM: C(MxN) = A(MxK,row) * Bt(NxK,row)^T --------------
// grid = (N/64, M/128): N-strips vary fastest so blocks sharing an A slab co-run
// (A slab stays in L2; HBM reads A once). Block = 128 threads (4 waves); each wave
// owns a 32x64 tile: 2 M-tiles x 4 N-tiles = 8 accumulators, B-frags reused across
// the two M-tiles (12 b128 loads per 8 WMMAs).
__global__ void __launch_bounds__(128)
egnn_gemm_bf16_wmma(const bf16* __restrict__ A, const bf16* __restrict__ Bt,
                    bf16* __restrict__ C, int M, int N, int K) {
  int wave = threadIdx.x >> 5;
  int lane = threadIdx.x & 31;
  int m0 = blockIdx.y * 128 + wave * 32;   // wave: rows m0..m0+31
  int n0 = blockIdx.x * 64;                // cols n0..n0+63
  int r16 = lane & 15;                     // A: row in tile / B: col in tile
  int khalf = (lane >> 4) * 16;            // lanes 16-31 carry K+16..K+31

  v8f acc00 = {}, acc01 = {}, acc02 = {}, acc03 = {};
  v8f acc10 = {}, acc11 = {}, acc12 = {}, acc13 = {};
  const bf16* ap0 = A + (size_t)(m0 + r16) * K + khalf;
  const bf16* ap1 = ap0 + (size_t)16 * K;
  const bf16* bp  = Bt + (size_t)(n0 + r16) * K + khalf;
  const size_t bs = (size_t)16 * K;

  for (int k = 0; k < K; k += 32) {
    __builtin_prefetch(ap0 + k + 32, 0, 1);  // global_prefetch_b8, next K-slab
    __builtin_prefetch(ap1 + k + 32, 0, 1);
    v16bf a0 = *(const v16bf*)(ap0 + k);
    v16bf a1 = *(const v16bf*)(ap1 + k);
    v16bf b0 = *(const v16bf*)(bp + k);
    v16bf b1 = *(const v16bf*)(bp + bs + k);
    v16bf b2 = *(const v16bf*)(bp + 2 * bs + k);
    v16bf b3 = *(const v16bf*)(bp + 3 * bs + k);
    acc00 = __builtin_amdgcn_wmma_f32_16x16x32_bf16(false, a0, false, b0, (short)0, acc00, false, false);
    acc10 = __builtin_amdgcn_wmma_f32_16x16x32_bf16(false, a1, false, b0, (short)0, acc10, false, false);
    acc01 = __builtin_amdgcn_wmma_f32_16x16x32_bf16(false, a0, false, b1, (short)0, acc01, false, false);
    acc11 = __builtin_amdgcn_wmma_f32_16x16x32_bf16(false, a1, false, b1, (short)0, acc11, false, false);
    acc02 = __builtin_amdgcn_wmma_f32_16x16x32_bf16(false, a0, false, b2, (short)0, acc02, false, false);
    acc12 = __builtin_amdgcn_wmma_f32_16x16x32_bf16(false, a1, false, b2, (short)0, acc12, false, false);
    acc03 = __builtin_amdgcn_wmma_f32_16x16x32_bf16(false, a0, false, b3, (short)0, acc03, false, false);
    acc13 = __builtin_amdgcn_wmma_f32_16x16x32_bf16(false, a1, false, b3, (short)0, acc13, false, false);
  }

  // C/D f32 layout: col = lane&15; row = vgpr + 8*(lane>=16)
  int col = lane & 15;
  int rbase = (lane >> 4) * 8;
  bf16* cp0 = C + (size_t)m0 * N + n0;
  bf16* cp1 = cp0 + (size_t)16 * N;
#pragma unroll
  for (int r = 0; r < 8; ++r) {
    size_t row = (size_t)(r + rbase) * N;
    cp0[row + col]      = (bf16)acc00[r];
    cp0[row + col + 16] = (bf16)acc01[r];
    cp0[row + col + 32] = (bf16)acc02[r];
    cp0[row + col + 48] = (bf16)acc03[r];
    cp1[row + col]      = (bf16)acc10[r];
    cp1[row + col + 16] = (bf16)acc11[r];
    cp1[row + col + 32] = (bf16)acc12[r];
    cp1[row + col + 48] = (bf16)acc13[r];
  }
}

// ------- fused: gather-sum over edges, /deg, +bias, LayerNorm, GELU, +residual ------
__global__ void egnn_agg_ln_gelu_kernel(const bf16* __restrict__ xt,
                                        const bf16* __restrict__ res,
                                        bf16* __restrict__ dst,
                                        const float* __restrict__ cb,
                                        const float* __restrict__ lns,
                                        const float* __restrict__ lnb,
                                        const int* __restrict__ row_start,
                                        const int* __restrict__ nbr,
                                        const float* __restrict__ invdeg,
                                        int fo) {
  __shared__ float sdata[256];
  int n = blockIdx.x, b = blockIdx.y, f = threadIdx.x, nt = blockDim.x;
  size_t rowbase = (size_t)b * NODES;

  float s = 0.f;
  int e0 = row_start[n], e1 = row_start[n + 1];
  for (int e = e0; e < e1; ++e) {
    int d = nbr[e];
    s += (float)xt[(rowbase + d) * fo + f];
  }
  float val = s * invdeg[n] + cb[f];

  sdata[f] = val; __syncthreads();
  for (int st = nt >> 1; st > 0; st >>= 1) { if (f < st) sdata[f] += sdata[f + st]; __syncthreads(); }
  float mean = sdata[0] / (float)nt; __syncthreads();
  float dv = val - mean;
  sdata[f] = dv * dv; __syncthreads();
  for (int st = nt >> 1; st > 0; st >>= 1) { if (f < st) sdata[f] += sdata[f + st]; __syncthreads(); }
  float var = sdata[0] / (float)nt; __syncthreads();

  float y = dv * rsqrtf(var + 1e-5f) * lns[f] + lnb[f];
  float out = gelu_exact(y) + (float)res[(rowbase + n) * fo + f];
  dst[(rowbase + n) * fo + f] = (bf16)out;
}

// ---------------- mean pool over nodes: (B,N,128) -> (B,128) -----------------------
__global__ void egnn_pool_kernel(const bf16* __restrict__ h, float* __restrict__ pooled) {
  int b = blockIdx.x, f = threadIdx.x;  // 128 threads
  size_t base = (size_t)b * NODES * 128 + f;
  float s = 0.f;
  for (int n = 0; n < NODES; ++n) s += (float)h[base + (size_t)n * 128];
  pooled[b * 128 + f] = s * (1.0f / (float)NODES);
}

// ---------------- classifier head: 128 -> 64 (LN,GELU) -> 7 ------------------------
__global__ void egnn_head_kernel(const float* __restrict__ pooled,
                                 const float* __restrict__ w1, const float* __restrict__ b1,
                                 const float* __restrict__ lns, const float* __restrict__ lnb,
                                 const float* __restrict__ w2, const float* __restrict__ b2,
                                 float* __restrict__ out) {
  __shared__ float pv[128];
  __shared__ float red[64];
  __shared__ float zs[64];
  int b = blockIdx.x, j = threadIdx.x;  // 64 threads
  pv[j] = pooled[b * 128 + j];
  pv[j + 64] = pooled[b * 128 + 64 + j];
  __syncthreads();
  float acc = b1[j];
  for (int k = 0; k < 128; ++k) acc += pv[k] * w1[k * 64 + j];
  red[j] = acc; __syncthreads();
  for (int st = 32; st > 0; st >>= 1) { if (j < st) red[j] += red[j + st]; __syncthreads(); }
  float mean = red[0] / 64.f; __syncthreads();
  float dv = acc - mean;
  red[j] = dv * dv; __syncthreads();
  for (int st = 32; st > 0; st >>= 1) { if (j < st) red[j] += red[j + st]; __syncthreads(); }
  float var = red[0] / 64.f; __syncthreads();
  float y = dv * rsqrtf(var + 1e-5f) * lns[j] + lnb[j];
  zs[j] = gelu_exact(y);
  __syncthreads();
  if (j < 7) {
    float o = b2[j];
    for (int t = 0; t < 64; ++t) o += zs[t] * w2[t * 7 + j];
    out[b * 7 + j] = o;
  }
}

extern "C" void kernel_launch(void* const* d_in, const int* in_sizes, int n_in,
                              void* d_out, int out_size, void* d_ws, size_t ws_size,
                              hipStream_t stream) {
  if (n_in < 26) return;
  // dict-insertion-order flattening (None leaves dropped)
  const float* x    = (const float*)d_in[0];
  const int*   ei   = (const int*)d_in[1];
  const float* ipw  = (const float*)d_in[2];
  const float* ipb  = (const float*)d_in[3];
  const float* cw0  = (const float*)d_in[4];
  const float* cb0  = (const float*)d_in[5];
  const float* ln0s = (const float*)d_in[6];
  const float* ln0b = (const float*)d_in[7];
  const float* pw0  = (const float*)d_in[8];
  const float* pb0  = (const float*)d_in[9];   (void)pb0;  // zeros
  const float* cw1  = (const float*)d_in[10];
  const float* cb1  = (const float*)d_in[11];
  const float* ln1s = (const float*)d_in[12];
  const float* ln1b = (const float*)d_in[13];
  const float* cw2  = (const float*)d_in[14];
  const float* cb2  = (const float*)d_in[15];
  const float* ln2s = (const float*)d_in[16];
  const float* ln2b = (const float*)d_in[17];
  const float* pw2  = (const float*)d_in[18];
  const float* pb2  = (const float*)d_in[19];  (void)pb2;  // zeros
  const float* cw_1 = (const float*)d_in[20];
  const float* cb_1 = (const float*)d_in[21];
  const float* clns = (const float*)d_in[22];
  const float* clnb = (const float*)d_in[23];
  const float* cw_2 = (const float*)d_in[24];
  const float* cb_2 = (const float*)d_in[25];
  int E = in_sizes[1] / 2;

  const size_t MN = (size_t)BATCH * NODES;  // 489472 = 3824 * 128
  char* p = (char*)d_ws;
  auto alloc = [&](size_t bytes) -> char* {
    char* r = p; p += (bytes + 255) & ~(size_t)255; return r;
  };
  bf16* bufA = (bf16*)alloc(MN * 256 * sizeof(bf16));
  bf16* bufB = (bf16*)alloc(MN * 256 * sizeof(bf16));
  bf16* xt   = (bf16*)alloc(MN * 256 * sizeof(bf16));
  bf16* wtc0 = (bf16*)alloc(128 * 256 * sizeof(bf16));
  bf16* wtp0 = (bf16*)alloc(128 * 256 * sizeof(bf16));
  bf16* wtc1 = (bf16*)alloc(256 * 256 * sizeof(bf16));
  bf16* wtc2 = (bf16*)alloc(256 * 128 * sizeof(bf16));
  bf16* wtp2 = (bf16*)alloc(256 * 128 * sizeof(bf16));
  float* pooled   = (float*)alloc(BATCH * 128 * sizeof(float));
  int*   deg      = (int*)alloc(512 * sizeof(int));
  int*   cursor   = (int*)alloc(512 * sizeof(int));
  int*   rowstart = (int*)alloc(512 * sizeof(int));
  int*   nbr      = (int*)alloc(8192 * sizeof(int));
  float* invdeg   = (float*)alloc(512 * sizeof(float));
  if ((size_t)(p - (char*)d_ws) > ws_size) return;

  // weight prep (bf16 transposed) + CSR
  egnn_wconv_kernel<<<(128 * 256 + 255) / 256, 256, 0, stream>>>(cw0, wtc0, 128, 256);
  egnn_wconv_kernel<<<(128 * 256 + 255) / 256, 256, 0, stream>>>(pw0, wtp0, 128, 256);
  egnn_wconv_kernel<<<(256 * 256 + 255) / 256, 256, 0, stream>>>(cw1, wtc1, 256, 256);
  egnn_wconv_kernel<<<(256 * 128 + 255) / 256, 256, 0, stream>>>(cw2, wtc2, 256, 128);
  egnn_wconv_kernel<<<(256 * 128 + 255) / 256, 256, 0, stream>>>(pw2, wtp2, 256, 128);
  egnn_csr_kernel<<<1, 512, 0, stream>>>(ei, E, deg, cursor, rowstart, nbr, invdeg);

  // input projection -> bufA (stride 128)
  egnn_inproj_kernel<<<dim3(NODES, BATCH), 128, 0, stream>>>(x, ipw, ipb, bufA);

  const int M = (int)MN;
  const int MB = M / 128;  // 3824 macro-row blocks
  dim3 gblk(128);
  // ---- block 0: fi=128, fo=256, with proj ----
  egnn_gemm_bf16_wmma<<<dim3(4, MB), gblk, 0, stream>>>(bufA, wtp0, bufB, M, 256, 128);
  egnn_gemm_bf16_wmma<<<dim3(4, MB), gblk, 0, stream>>>(bufA, wtc0, xt,   M, 256, 128);
  egnn_agg_ln_gelu_kernel<<<dim3(NODES, BATCH), 256, 0, stream>>>(
      xt, bufB, bufB, cb0, ln0s, ln0b, rowstart, nbr, invdeg, 256);
  // ---- block 1: fi=256, fo=256, no proj (residual = h, in place) ----
  egnn_gemm_bf16_wmma<<<dim3(4, MB), gblk, 0, stream>>>(bufB, wtc1, xt, M, 256, 256);
  egnn_agg_ln_gelu_kernel<<<dim3(NODES, BATCH), 256, 0, stream>>>(
      xt, bufB, bufB, cb1, ln1s, ln1b, rowstart, nbr, invdeg, 256);
  // ---- block 2: fi=256, fo=128, with proj ----
  egnn_gemm_bf16_wmma<<<dim3(2, MB), gblk, 0, stream>>>(bufB, wtp2, bufA, M, 128, 256);
  egnn_gemm_bf16_wmma<<<dim3(2, MB), gblk, 0, stream>>>(bufB, wtc2, xt,   M, 128, 256);
  egnn_agg_ln_gelu_kernel<<<dim3(NODES, BATCH), 128, 0, stream>>>(
      xt, bufA, bufA, cb2, ln2s, ln2b, rowstart, nbr, invdeg, 128);

  // pool + head
  egnn_pool_kernel<<<BATCH, 128, 0, stream>>>(bufA, pooled);
  egnn_head_kernel<<<BATCH, 64, 0, stream>>>(pooled, cw_1, cb_1, clns, clnb, cw_2, cb_2,
                                             (float*)d_out);
}
